// TriangleAttention_67611375173895
// MI455X (gfx1250) — matrix operand order repaired and checked by reading, not compiled, not run on hardware
//
#include <hip/hip_runtime.h>

#define IDIM 384
#define JDIM 384
#define CDIM 128
#define HEADS 4
#define DHD 32
#define FDIM 128
#define NPOS (IDIM * JDIM)
#define INFV 1e9f
#define EPSV 1e-5f

typedef __attribute__((ext_vector_type(16))) _Float16 v16h;
typedef __attribute__((ext_vector_type(8)))  _Float16 v8h;
typedef __attribute__((ext_vector_type(4)))  _Float16 v4h;
typedef __attribute__((ext_vector_type(8)))  float    v8f;
typedef __attribute__((ext_vector_type(4)))  float    v4f;

union V16U { v16h v; v8h h[2]; };

#if __has_builtin(__builtin_amdgcn_global_load_async_to_lds_b128) && \
    __has_builtin(__builtin_amdgcn_global_store_async_from_lds_b128)
#define HAVE_ASYNC_LDS 1
typedef int v4i __attribute__((vector_size(4 * sizeof(int))));
typedef __attribute__((address_space(1))) v4i gas_b128;  // global int4
typedef __attribute__((address_space(3))) v4i las_b128;  // LDS int4
#else
#define HAVE_ASYNC_LDS 0
#endif

__device__ __forceinline__ v8f wmma_f16(v16h a, v16h b, v8f c) {
  // D = A(16x32 f16) * B(32x16 f16) + C(16x16 f32)
  return __builtin_amdgcn_wmma_f32_16x16x32_f16(false, a, false, b, (short)0, c,
                                                false, false);
}

__device__ __forceinline__ float wsum32(float v) {
#pragma unroll
  for (int m = 1; m < 32; m <<= 1) v += __shfl_xor(v, m, 32);
  return v;
}

// -------------------------------------------------------------------------
// Kernel 1: fused LayerNorm + tri bias + {q,k,v,g} projections (f16 WMMA).
// Block = 48 positions (one i-row slice of 48 j's). 256 threads = 8 waves.
// -------------------------------------------------------------------------
__global__ __launch_bounds__(256) void k_ln_proj(
    const float* __restrict__ x, const float* __restrict__ ln_w,
    const float* __restrict__ ln_b, const float* __restrict__ w_tri,
    const float* __restrict__ wq, const float* __restrict__ wk,
    const float* __restrict__ wv, const float* __restrict__ wg,
    const float* __restrict__ bg, _Float16* __restrict__ qout,
    _Float16* __restrict__ kout, _Float16* __restrict__ vT,
    _Float16* __restrict__ gout, float* __restrict__ tri) {
  __shared__ alignas(16) _Float16 xn[48][136];     // LN'd activations (f16)
  __shared__ alignas(16) _Float16 wT[128][136];    // weight transposed [f][c]
  __shared__ alignas(16) _Float16 stage[48][136];  // row-major store stage

  const int tid = threadIdx.x, lane = tid & 31, wave = tid >> 5;
  const int p0 = (int)blockIdx.x * 48;
  const int i = p0 / JDIM, j0 = p0 % JDIM;

  // prefetch weights into L2/WGP$ (global_prefetch_b8), 256B per thread
  __builtin_prefetch(wq + tid * 64, 0, 1);
  __builtin_prefetch(wk + tid * 64, 0, 1);
  __builtin_prefetch(wv + tid * 64, 0, 1);
  __builtin_prefetch(wg + tid * 64, 0, 1);

  // ---------------- Phase A: LayerNorm + triangle bias ----------------
  const int c4 = lane * 4;
  v4f lw = *(const v4f*)(ln_w + c4);
  v4f lb = *(const v4f*)(ln_b + c4);
  v4f wt0 = *(const v4f*)(w_tri + (size_t)(c4 + 0) * HEADS);
  v4f wt1 = *(const v4f*)(w_tri + (size_t)(c4 + 1) * HEADS);
  v4f wt2 = *(const v4f*)(w_tri + (size_t)(c4 + 2) * HEADS);
  v4f wt3 = *(const v4f*)(w_tri + (size_t)(c4 + 3) * HEADS);

#pragma unroll
  for (int rr = 0; rr < 6; ++rr) {
    const int jl = wave * 6 + rr;
    v4f xv = *(const v4f*)(x + (size_t)(p0 + jl) * CDIM + c4);
    float s = xv.x + xv.y + xv.z + xv.w;
    float sq = xv.x * xv.x + xv.y * xv.y + xv.z * xv.z + xv.w * xv.w;
    s = wsum32(s);
    sq = wsum32(sq);
    const float mu = s * (1.0f / 128.0f);
    const float var = sq * (1.0f / 128.0f) - mu * mu;
    const float rstd = rsqrtf(var + EPSV);
    v4f xf;
    xf.x = (xv.x - mu) * rstd * lw.x + lb.x;
    xf.y = (xv.y - mu) * rstd * lw.y + lb.y;
    xf.z = (xv.z - mu) * rstd * lw.z + lb.z;
    xf.w = (xv.w - mu) * rstd * lw.w + lb.w;
    v4h xh;
    xh.x = (_Float16)xf.x; xh.y = (_Float16)xf.y;
    xh.z = (_Float16)xf.z; xh.w = (_Float16)xf.w;
    *(v4h*)&xn[jl][c4] = xh;
    // tri[h] = sum_c xn * w_tri[c][h]
    float t0 = xf.x * wt0.x + xf.y * wt1.x + xf.z * wt2.x + xf.w * wt3.x;
    float t1 = xf.x * wt0.y + xf.y * wt1.y + xf.z * wt2.y + xf.w * wt3.y;
    float t2 = xf.x * wt0.z + xf.y * wt1.z + xf.z * wt2.z + xf.w * wt3.z;
    float t3 = xf.x * wt0.w + xf.y * wt1.w + xf.z * wt2.w + xf.w * wt3.w;
    t0 = wsum32(t0); t1 = wsum32(t1); t2 = wsum32(t2); t3 = wsum32(t3);
    if (lane == 0) {
      float* tp = tri + (size_t)i * HEADS * JDIM + (j0 + jl);
      tp[0 * JDIM] = t0; tp[1 * JDIM] = t1;
      tp[2 * JDIM] = t2; tp[3 * JDIM] = t3;
    }
  }

  // ---------------- Phase B: 4 projections via WMMA ----------------
  const float* wmats[4] = {wq, wk, wv, wg};
  for (int mat = 0; mat < 4; ++mat) {
    const float* w = wmats[mat];
    // cooperative transpose-load weights to LDS (f16)
#pragma unroll
    for (int t = 0; t < 16; ++t) {
      int idx = tid + t * 256;  // 4096 float4 = 128x128 f32
      int c = idx >> 5;
      int f4 = (idx & 31) * 4;
      v4f w4 = *(const v4f*)(w + (size_t)c * FDIM + f4);
      wT[f4 + 0][c] = (_Float16)w4.x;
      wT[f4 + 1][c] = (_Float16)w4.y;
      wT[f4 + 2][c] = (_Float16)w4.z;
      wT[f4 + 3][c] = (_Float16)w4.w;
    }
    __syncthreads();

    const int koff = (lane < 16) ? 0 : 8;
    const int krun = (lane >> 4) * 16;
    const int rbase = 8 * (lane >> 4);
#pragma unroll
    for (int t = 0; t < 3; ++t) {
      const int tile = wave * 3 + t;  // 24 tiles: 3 Mt x 8 Nt
      const int Mt = tile >> 3, Nt = tile & 7;
      const int arow = Mt * 16 + (lane & 15);
      const int bn = Nt * 16 + (lane & 15);
      v8f acc = {};
#pragma unroll
      for (int ks = 0; ks < 4; ++ks) {
        V16U a, b;
        a.h[0] = *(const v8h*)&xn[arow][ks * 32 + koff];
        a.h[1] = *(const v8h*)&xn[arow][ks * 32 + koff + 16];
        b.h[0] = *(const v8h*)&wT[bn][ks * 32 + koff];
        b.h[1] = *(const v8h*)&wT[bn][ks * 32 + koff + 16];
        acc = wmma_f16(a.v, b.v, acc);
      }
      if (mat == 2) {
        // v: store column-major vT[i][f][j] (per-lane contiguous in M)
        v8h pk;
#pragma unroll
        for (int r = 0; r < 8; ++r) pk[r] = (_Float16)acc[r];
        size_t off = ((size_t)i * FDIM + bn) * JDIM + j0 + Mt * 16 + rbase;
        *(v8h*)(vT + off) = pk;
      } else {
        const float qs = (mat == 0) ? 0.17677669529663687f : 1.0f;  // 1/sqrt(32)
        const float bb = (mat == 3) ? bg[bn] : 0.0f;
#pragma unroll
        for (int r = 0; r < 8; ++r) {
          float vv = acc[r] * qs;
          if (mat == 3) vv = 1.0f / (1.0f + __expf(-(vv + bb)));
          stage[Mt * 16 + rbase + r][bn] = (_Float16)vv;
        }
      }
    }
    __syncthreads();
    if (mat != 2) {
      _Float16* dst = (mat == 0) ? qout : ((mat == 1) ? kout : gout);
#pragma unroll
      for (int t = 0; t < 3; ++t) {
        int idx = tid + t * 256;  // 768 chunks of 8 halves
        int row = idx >> 4;
        int c8 = (idx & 15) * 8;
        _Float16* gp = dst + (size_t)(p0 + row) * FDIM + c8;
#if HAVE_ASYNC_LDS
        // async LDS -> global store, bypasses VGPRs (ASYNCcnt tracked)
        __builtin_amdgcn_global_store_async_from_lds_b128(
            (gas_b128*)gp, (las_b128*)&stage[row][c8], 0, 0);
#else
        *(v8h*)gp = *(const v8h*)&stage[row][c8];
#endif
      }
#if HAVE_ASYNC_LDS
      asm volatile("s_wait_asynccnt 0" ::: "memory");
#endif
    }
    __syncthreads();
  }
}

// -------------------------------------------------------------------------
// Kernel 2: attention for one (i, h). Flash-style online softmax, all WMMA.
// K-slice + vT-slice + bias live in LDS; P repacked via per-wave LDS stage.
// -------------------------------------------------------------------------
__global__ __launch_bounds__(256) void k_attn(
    const _Float16* __restrict__ qin, const _Float16* __restrict__ kin,
    const _Float16* __restrict__ vT, const _Float16* __restrict__ gin,
    const float* __restrict__ tri, const float* __restrict__ mask,
    _Float16* __restrict__ og) {
  __shared__ alignas(16) _Float16 ksl[384][40];   // k[j][d] slice for head h
  __shared__ alignas(16) _Float16 vsl[32][384];   // vT[d][j] slice
  __shared__ alignas(16) float bias[384];         // tri + mask bias per key
  __shared__ alignas(16) _Float16 pst[8][16][32]; // per-wave P/O repack stage

  const int tid = threadIdx.x, lane = tid & 31, wave = tid >> 5;
  const int i = (int)blockIdx.x, h = (int)blockIdx.y;

#pragma unroll
  for (int t = 0; t < 6; ++t) {  // 1536 16B chunks: k slice
    int idx = tid + t * 256;
    int row = idx >> 2, c8 = (idx & 3) * 8;
    const _Float16* src =
        kin + (size_t)(i * JDIM + row) * FDIM + h * DHD + c8;
#if HAVE_ASYNC_LDS
    __builtin_amdgcn_global_load_async_to_lds_b128(
        (gas_b128*)src, (las_b128*)&ksl[row][c8], 0, 0);
#else
    *(v8h*)&ksl[row][c8] = *(const v8h*)src;
#endif
  }
#pragma unroll
  for (int t = 0; t < 6; ++t) {  // 1536 16B chunks: vT slice
    int idx = tid + t * 256;
    int row = idx / 48, cc = (idx % 48) * 8;
    const _Float16* src = vT + ((size_t)i * FDIM + h * DHD + row) * JDIM + cc;
#if HAVE_ASYNC_LDS
    __builtin_amdgcn_global_load_async_to_lds_b128(
        (gas_b128*)src, (las_b128*)&vsl[row][cc], 0, 0);
#else
    *(v8h*)&vsl[row][cc] = *(const v8h*)src;
#endif
  }
  for (int jj = tid; jj < 384; jj += 256)
    bias[jj] = tri[((size_t)i * HEADS + h) * JDIM + jj] +
               INFV * (mask[(size_t)i * JDIM + jj] - 1.0f);
#if HAVE_ASYNC_LDS
  asm volatile("s_wait_asynccnt 0" ::: "memory");
#endif
  __syncthreads();

  const int bn = lane & 15;
  const int krun = (lane >> 4) * 16;
  const int koff = (lane < 16) ? 0 : 8;
  const int rbase = 8 * (lane >> 4);

  for (int qt = wave; qt < 24; qt += 8) {
    const int q0 = qt * 16;
    V16U a;
    {
      const _Float16* qp =
          qin + (size_t)(i * JDIM + q0 + (lane & 15)) * FDIM + h * DHD;
      a.h[0] = *(const v8h*)(qp + koff);
      a.h[1] = *(const v8h*)(qp + koff + 16);
    }
    float m[8], lsum[8];
    v8f o0 = {}, o1 = {};
#pragma unroll
    for (int r = 0; r < 8; ++r) { m[r] = -INFINITY; lsum[r] = 0.0f; }

    for (int kp = 0; kp < 12; ++kp) {  // 12 pairs of 16-key tiles
      const int k0 = kp * 32;
      V16U b0, b1;
      b0.h[0] = *(const v8h*)&ksl[k0 + bn][koff];
      b0.h[1] = *(const v8h*)&ksl[k0 + bn][koff + 16];
      b1.h[0] = *(const v8h*)&ksl[k0 + 16 + bn][koff];
      b1.h[1] = *(const v8h*)&ksl[k0 + 16 + bn][koff + 16];
      v8f zero = {};
      v8f s0 = wmma_f16(a.v, b0.v, zero);
      v8f s1 = wmma_f16(a.v, b1.v, zero);
      const float bb0 = bias[k0 + bn], bb1 = bias[k0 + 16 + bn];
      float e0[8], e1[8], sc[8];
#pragma unroll
      for (int r = 0; r < 8; ++r) {
        float v0 = s0[r] + bb0, v1 = s1[r] + bb1;
        float mx = fmaxf(v0, v1);
        mx = fmaxf(mx, __shfl_xor(mx, 1, 32));
        mx = fmaxf(mx, __shfl_xor(mx, 2, 32));
        mx = fmaxf(mx, __shfl_xor(mx, 4, 32));
        mx = fmaxf(mx, __shfl_xor(mx, 8, 32));
        const float mn = fmaxf(m[r], mx);
        const float scale = __expf(m[r] - mn);
        const float p0v = __expf(v0 - mn), p1v = __expf(v1 - mn);
        float rs = p0v + p1v;
        rs += __shfl_xor(rs, 1, 32);
        rs += __shfl_xor(rs, 2, 32);
        rs += __shfl_xor(rs, 4, 32);
        rs += __shfl_xor(rs, 8, 32);
        lsum[r] = lsum[r] * scale + rs;
        m[r] = mn;
        sc[r] = scale; e0[r] = p0v; e1[r] = p1v;
      }
#pragma unroll
      for (int r = 0; r < 8; ++r) { o0[r] *= sc[r]; o1[r] *= sc[r]; }
      // repack P (C-layout) -> A-layout via per-wave LDS stage
#pragma unroll
      for (int r = 0; r < 8; ++r) {
        pst[wave][rbase + r][bn] = (_Float16)e0[r];
        pst[wave][rbase + r][16 + bn] = (_Float16)e1[r];
      }
      asm volatile("s_wait_dscnt 0" ::: "memory");
      V16U pf, bv0, bv1;
      pf.h[0] = *(const v8h*)&pst[wave][lane & 15][koff];
      pf.h[1] = *(const v8h*)&pst[wave][lane & 15][koff + 16];
      bv0.h[0] = *(const v8h*)&vsl[bn][k0 + koff];
      bv0.h[1] = *(const v8h*)&vsl[bn][k0 + koff + 16];
      bv1.h[0] = *(const v8h*)&vsl[16 + bn][k0 + koff];
      bv1.h[1] = *(const v8h*)&vsl[16 + bn][k0 + koff + 16];
      o0 = wmma_f16(pf.v, bv0.v, o0);
      o1 = wmma_f16(pf.v, bv1.v, o1);
    }
    // epilogue: normalize, gate, store og (f16)
#pragma unroll
    for (int r = 0; r < 8; ++r) {
      const float inv = 1.0f / lsum[r];
      pst[wave][rbase + r][bn] = (_Float16)(o0[r] * inv);
      pst[wave][rbase + r][16 + bn] = (_Float16)(o1[r] * inv);
    }
    asm volatile("s_wait_dscnt 0" ::: "memory");
#pragma unroll
    for (int t = 0; t < 2; ++t) {
      int idx = lane + t * 32;  // 64 chunks of 8 halves
      int row = idx >> 2, cc = (idx & 3) * 8;
      v8h ov = *(const v8h*)&pst[wave][row][cc];
      size_t goff = (size_t)(i * JDIM + q0 + row) * FDIM + h * DHD + cc;
      v8h gv = *(const v8h*)(gin + goff);
      v8h res = ov * gv;  // v_pk_mul_f16
      *(v8h*)(og + goff) = res;
    }
  }
}

// -------------------------------------------------------------------------
// Kernel 3: output projection out = og @ wo + bo (f32 out), WMMA.
// -------------------------------------------------------------------------
__global__ __launch_bounds__(256) void k_out(const _Float16* __restrict__ og,
                                             const float* __restrict__ wo,
                                             const float* __restrict__ bo,
                                             float* __restrict__ out) {
  __shared__ alignas(16) _Float16 wT[128][136];  // wo transposed [c][f] f16
  __shared__ alignas(16) float stage[48][132];

  const int tid = threadIdx.x, lane = tid & 31, wave = tid >> 5;
  const int p0 = (int)blockIdx.x * 48;

  __builtin_prefetch(wo + tid * 64, 0, 1);

#pragma unroll
  for (int t = 0; t < 16; ++t) {
    int idx = tid + t * 256;  // 4096 float4
    int f = idx >> 5;
    int c4 = (idx & 31) * 4;
    v4f w4 = *(const v4f*)(wo + (size_t)f * CDIM + c4);
    wT[c4 + 0][f] = (_Float16)w4.x;
    wT[c4 + 1][f] = (_Float16)w4.y;
    wT[c4 + 2][f] = (_Float16)w4.z;
    wT[c4 + 3][f] = (_Float16)w4.w;
  }
  __syncthreads();

  const int koff = (lane < 16) ? 0 : 8;
  const int krun = (lane >> 4) * 16;
  const int rbase = 8 * (lane >> 4);
#pragma unroll
  for (int t = 0; t < 3; ++t) {
    const int tile = wave * 3 + t;  // 3 Mt x 8 Nt
    const int Mt = tile >> 3, Nt = tile & 7;
    const int bn = Nt * 16 + (lane & 15);
    const _Float16* ap = og + (size_t)(p0 + Mt * 16 + (lane & 15)) * FDIM;
    v8f acc = {};
#pragma unroll
    for (int ks = 0; ks < 4; ++ks) {
      V16U a, b;
      a.h[0] = *(const v8h*)(ap + ks * 32 + koff);
      a.h[1] = *(const v8h*)(ap + ks * 32 + koff + 16);
      b.h[0] = *(const v8h*)&wT[bn][ks * 32 + koff];
      b.h[1] = *(const v8h*)&wT[bn][ks * 32 + koff + 16];
      acc = wmma_f16(a.v, b.v, acc);
    }
    const float bv = bo[bn];
#pragma unroll
    for (int r = 0; r < 8; ++r) stage[Mt * 16 + rbase + r][bn] = acc[r] + bv;
  }
  __syncthreads();
#pragma unroll
  for (int t = 0; t < 6; ++t) {
    int idx = tid + t * 256;  // 1536 float4 chunks
    int row = idx >> 5, c4 = (idx & 31) * 4;
    *(v4f*)(out + (size_t)(p0 + row) * CDIM + c4) =
        *(const v4f*)&stage[row][c4];
  }
}

// -------------------------------------------------------------------------
extern "C" void kernel_launch(void* const* d_in, const int* in_sizes, int n_in,
                              void* d_out, int out_size, void* d_ws,
                              size_t ws_size, hipStream_t stream) {
  const float* x    = (const float*)d_in[0];
  const float* mask = (const float*)d_in[1];
  const float* ln_w = (const float*)d_in[2];
  const float* ln_b = (const float*)d_in[3];
  const float* wtri = (const float*)d_in[4];
  const float* wq   = (const float*)d_in[5];
  const float* wk   = (const float*)d_in[6];
  const float* wv   = (const float*)d_in[7];
  const float* wg   = (const float*)d_in[8];
  const float* bg   = (const float*)d_in[9];
  const float* wo   = (const float*)d_in[10];
  const float* bo   = (const float*)d_in[11];
  float* out = (float*)d_out;

  const size_t NPF = (size_t)NPOS * FDIM;
  _Float16* q  = (_Float16*)d_ws;
  _Float16* k  = q + NPF;
  _Float16* vT = k + NPF;
  _Float16* g  = vT + NPF;
  _Float16* og = g + NPF;
  float* tri = (float*)(og + NPF);

  k_ln_proj<<<NPOS / 48, 256, 0, stream>>>(x, ln_w, ln_b, wtri, wq, wk, wv, wg,
                                           bg, q, k, vT, g, tri);
  k_attn<<<dim3(IDIM, HEADS), 256, 0, stream>>>(q, k, vT, g, tri, mask, og);
  k_out<<<NPOS / 48, 256, 0, stream>>>(og, wo, bo, out);
}
